// PrimeLayerCaps_34626026340527
// MI455X (gfx1250) — matrix operand order repaired and checked by reading, not compiled, not run on hardware
//
#include <hip/hip_runtime.h>
#include <hip/hip_bf16.h>

// ---------------------------------------------------------------------------
// PrimaryCaps conv (256->256, 9x9, stride 2) as implicit GEMM, bf16x3 split
// precision on v_wmma_f32_16x16x32_bf16 with fp32 accumulation.
//
//  K-order: k = (kh*9+kw)*256 + ic  (spatial-major), K = 20736 = 648*32
//  N      : b*36 + oh*6 + ow,       N = 18432 = 288*64
//  Prep   : W split once into bf16 hi/lo planes in d_ws, laid out [oc][s][ic]
//           so A-tiles are contiguous -> loaded by the Tensor Data Mover.
//  Block  : 256 thr (8 waves), tile 256M x 64N, K-chunk 32, double-buffered.
//  Wave   : 32M x 64N = 8 accumulators -> 24 WMMA per K-chunk.
// ---------------------------------------------------------------------------

typedef __attribute__((ext_vector_type(16))) __bf16        bf16x16;
typedef __attribute__((ext_vector_type(8)))  float         f32x8;
typedef __attribute__((ext_vector_type(4)))  unsigned int  u32x4;
typedef __attribute__((ext_vector_type(4)))  unsigned int  v4u;
typedef __attribute__((ext_vector_type(8)))  int           v8i;
typedef __attribute__((ext_vector_type(4)))  int           v4i;

#define K_TOT    20736
#define KSTEPS   648
#define W_ELEMS  5308416          // 256*20736 elements per bf16 plane
#define LD       40               // halves per LDS row (80 B: 16B-aligned, bank-safe)
#define Y_PER_B  9216

// LDS layout (single static block, offsets are absolute LDS offsets)
#define OFF_AHI  0
#define OFF_ALO  20480            // 256 rows * 80 B
#define OFF_BHI  40960
#define OFF_BLO  46080            // 64 rows * 80 B
#define BUFSZ    51200            // double buffered -> 102400 B total

__device__ __forceinline__ unsigned short bf16_rne(float x) {
    unsigned u = __builtin_bit_cast(unsigned, x);
    u += 0x7FFFu + ((u >> 16) & 1u);
    return (unsigned short)(u >> 16);
}
__device__ __forceinline__ float bf16_f32(unsigned short h) {
    return __builtin_bit_cast(float, (unsigned)h << 16);
}
__device__ __forceinline__ unsigned pack2(unsigned short a, unsigned short b) {
    return (unsigned)a | ((unsigned)b << 16);
}
__device__ __forceinline__ bf16x16 load_frag(const unsigned short* p, int off2_halves) {
    union { u32x4 u[2]; bf16x16 v; } x;
    x.u[0] = *(const u32x4*)(p);
    x.u[1] = *(const u32x4*)(p + off2_halves);
    return x.v;
}
__device__ __forceinline__ f32x8 wmma_bf16(bf16x16 a, bf16x16 b, f32x8 c) {
    return __builtin_amdgcn_wmma_f32_16x16x32_bf16(false, a, false, b,
                                                   (short)0, c, false, false);
}

// TDM: 2-D tile load, 32 cols (bf16) x 256 rows, row stride 20736 halves,
// LDS gets 80 B row pitch via pad_interval=16 dwords, pad_amount=4 dwords.
__device__ __forceinline__ void tdm_load_a(unsigned lds_off, const unsigned short* g) {
    unsigned long long ga = (unsigned long long)(uintptr_t)g;
    v4u g0;
    g0[0] = 1u;                                    // count=1, user mode, load
    g0[1] = lds_off;                               // lds_addr [63:32]
    g0[2] = (unsigned)(ga & 0xFFFFFFFFu);          // global_addr lo
    g0[3] = (unsigned)((ga >> 32) & 0x01FFFFFFu)   // global_addr hi (57-bit)
          | 0x80000000u;                           // type=2 ("image")
    v8i g1;
    g1[0] = (1 << 16)                              // data_size = 2 B
          | (1 << 20)                              // pad_enable
          | (3 << 22)                              // pad_interval: 16 dwords (64 B rows)
          | (3 << 25);                             // pad_amount: 4 dwords (16 B)
    g1[1] = (int)((K_TOT & 0xFFFF) << 16);         // tensor_dim0[15:0]  @ [63:48]
    g1[2] = (int)(((K_TOT >> 16) & 0xFFFF)         // tensor_dim0[31:16] @ [79:64]
          | (256u << 16));                         // tensor_dim1[15:0]  @ [95:80]
    g1[3] = (int)(32u << 16);                      // tile_dim0 = 32     @ [127:112]
    g1[4] = 256;                                   // tile_dim1 = 256    @ [143:128]
    g1[5] = K_TOT;                                 // tensor_dim0_stride lo32
    g1[6] = 0;
    g1[7] = 0;
    v4i z4 = {0, 0, 0, 0};
    v8i z8 = {0, 0, 0, 0, 0, 0, 0, 0};
    __builtin_amdgcn_tensor_load_to_lds(g0, g1, z4, z4, z8, 0);
}

// ---- one-time prep: W [oc][ic][kh][kw] f32 -> hi/lo bf16 planes [oc][s][ic] ----
__global__ __launch_bounds__(256)
void prep_w(const float* __restrict__ W, unsigned short* __restrict__ Whi,
            unsigned short* __restrict__ Wlo)
{
    __shared__ unsigned int tb[20736];             // packed hi|lo<<16
    const int oc = blockIdx.x;
    const int t  = threadIdx.x;
    const float* src = W + (size_t)oc * K_TOT;
    for (int i = t; i < K_TOT; i += 256) {         // coalesced reads
        float x = src[i];
        unsigned short h = bf16_rne(x);
        unsigned short l = bf16_rne(x - bf16_f32(h));
        tb[i] = pack2(h, l);                       // index = ic*81 + s
    }
    __syncthreads();
    unsigned short* dh = Whi + (size_t)oc * K_TOT;
    unsigned short* dl = Wlo + (size_t)oc * K_TOT;
    for (int i = t; i < K_TOT; i += 256) {         // coalesced writes, i = s*256+ic
        int s  = i >> 8;
        int ic = i & 255;
        unsigned u = tb[ic * 81 + s];
        dh[i] = (unsigned short)u;
        dl[i] = (unsigned short)(u >> 16);
    }
}

// ---- main GEMM ----
__global__ __launch_bounds__(256)
void caps_conv_wmma(const float* __restrict__ X,
                    const unsigned short* __restrict__ Whi,
                    const unsigned short* __restrict__ Wlo,
                    const float* __restrict__ bias, float* __restrict__ out)
{
    __shared__ __align__(16) unsigned char smem[2 * BUFSZ];

    const int tid   = threadIdx.x;
    const int lane  = tid & 31;
    const int wv    = tid >> 5;          // wave 0..7 -> M offset wv*32
    const int lrow  = lane & 15;
    const int lhalf = lane >> 4;

    const int n_base = blockIdx.x * 64;  // 288 blocks

    // B staging: thread -> (n column 0..63, octet 0..3 of 32 ic values)
    const int b_nl   = tid & 63;
    const int b_koct = tid >> 6;
    const int n_g    = n_base + b_nl;
    const int nb     = n_g / 36;
    const int ps     = n_g - nb * 36;
    const int oh     = ps / 6;
    const int ow     = ps - oh * 6;
    const float* xptr0 = X + (size_t)nb * 102400 + (size_t)(2 * oh) * 20 + (2 * ow);

    f32x8 acc[2][4] = {};

    auto stageA = [&](int kc, int buf) {
        if (wv == 0) {
            tdm_load_a((unsigned)(buf * BUFSZ + OFF_AHI), Whi + (size_t)kc * 32);
            tdm_load_a((unsigned)(buf * BUFSZ + OFF_ALO), Wlo + (size_t)kc * 32);
        }
    };
    auto stageB = [&](int kc, int buf) {
        const int s   = kc >> 3;                  // uniform: which (kh,kw)
        const int icb = (kc & 7) * 32;
        const int kh  = s / 9;
        const int kw  = s - kh * 9;
        const float* p = xptr0 + (size_t)(icb + b_koct * 8) * 400 + kh * 20 + kw;
        float v[8];
        #pragma unroll
        for (int j = 0; j < 8; ++j) v[j] = p[(size_t)j * 400];   // imm offsets
        unsigned hu[4], lu[4];
        #pragma unroll
        for (int j = 0; j < 8; j += 2) {
            unsigned short h0 = bf16_rne(v[j]),     h1 = bf16_rne(v[j + 1]);
            unsigned short l0 = bf16_rne(v[j]     - bf16_f32(h0));
            unsigned short l1 = bf16_rne(v[j + 1] - bf16_f32(h1));
            hu[j >> 1] = pack2(h0, h1);
            lu[j >> 1] = pack2(l0, l1);
        }
        unsigned short* bh = (unsigned short*)(smem + buf * BUFSZ + OFF_BHI)
                           + b_nl * LD + b_koct * 8;
        unsigned short* bl = (unsigned short*)(smem + buf * BUFSZ + OFF_BLO)
                           + b_nl * LD + b_koct * 8;
        *(u32x4*)bh = (u32x4){hu[0], hu[1], hu[2], hu[3]};
        *(u32x4*)bl = (u32x4){lu[0], lu[1], lu[2], lu[3]};
    };

    // prologue: fill buffer 0
    stageA(0, 0);
    stageB(0, 0);
    if (wv == 0) __builtin_amdgcn_s_wait_tensorcnt(0);
    __syncthreads();

    int cur = 0;
    for (int kc = 0; kc < KSTEPS; ++kc) {
        const int nxt = cur ^ 1;
        if (kc + 1 < KSTEPS) {               // uniform branch
            stageA(kc + 1, nxt);
            stageB(kc + 1, nxt);
        }

        const unsigned short* Ah = (const unsigned short*)(smem + cur * BUFSZ + OFF_AHI);
        const unsigned short* Al = (const unsigned short*)(smem + cur * BUFSZ + OFF_ALO);
        const unsigned short* Bh = (const unsigned short*)(smem + cur * BUFSZ + OFF_BHI);
        const unsigned short* Bl = (const unsigned short*)(smem + cur * BUFSZ + OFF_BLO);

        bf16x16 ah[2], al[2], bh[4], bl[4];
        #pragma unroll
        for (int mi = 0; mi < 2; ++mi) {
            const int row = (wv * 32 + mi * 16 + lrow) * LD + lhalf * 8;
            ah[mi] = load_frag(Ah + row, 16);     // K {0..7,16..23} / {8..15,24..31}
            al[mi] = load_frag(Al + row, 16);
        }
        #pragma unroll
        for (int ni = 0; ni < 4; ++ni) {
            const int col = (ni * 16 + lrow) * LD + lhalf * 8;
            bh[ni] = load_frag(Bh + col, 16);     // K {0..7,16..23} / {8..15,24..31}
            bl[ni] = load_frag(Bl + col, 16);
        }

        // bf16x3, term-major -> independent WMMA chains
        #pragma unroll
        for (int mi = 0; mi < 2; ++mi)
            #pragma unroll
            for (int ni = 0; ni < 4; ++ni)
                acc[mi][ni] = wmma_bf16(ah[mi], bh[ni], acc[mi][ni]);
        #pragma unroll
        for (int mi = 0; mi < 2; ++mi)
            #pragma unroll
            for (int ni = 0; ni < 4; ++ni)
                acc[mi][ni] = wmma_bf16(ah[mi], bl[ni], acc[mi][ni]);
        #pragma unroll
        for (int mi = 0; mi < 2; ++mi)
            #pragma unroll
            for (int ni = 0; ni < 4; ++ni)
                acc[mi][ni] = wmma_bf16(al[mi], bh[ni], acc[mi][ni]);

        if (wv == 0) __builtin_amdgcn_s_wait_tensorcnt(0);
        __syncthreads();
        cur = nxt;
    }

    // epilogue: +bias, scatter to raw conv layout [b][oc*36+pos]
    #pragma unroll
    for (int mi = 0; mi < 2; ++mi) {
        #pragma unroll
        for (int ni = 0; ni < 4; ++ni) {
            const int n  = n_base + ni * 16 + lrow;
            const int bo = n / 36;
            const int pp = n - bo * 36;
            const size_t obase = (size_t)bo * Y_PER_B + pp;
            #pragma unroll
            for (int r = 0; r < 8; ++r) {
                const int oc = wv * 32 + mi * 16 + lhalf * 8 + r;
                out[obase + (size_t)oc * 36] = acc[mi][ni][r] + bias[oc];
            }
        }
    }
}

// squash over consecutive groups of 8 floats, in place on d_out.
__global__ __launch_bounds__(256)
void caps_squash(float* __restrict__ y)
{
    const size_t g = (size_t)blockIdx.x * blockDim.x + threadIdx.x;
    float4* p = (float4*)y + g * 2;
    float4 a = p[0];
    float4 b = p[1];
    float ns = a.x * a.x + a.y * a.y + a.z * a.z + a.w * a.w
             + b.x * b.x + b.y * b.y + b.z * b.z + b.w * b.w;
    float s = (ns / (1.0f + ns)) / (0.001f + sqrtf(ns));
    a.x *= s; a.y *= s; a.z *= s; a.w *= s;
    b.x *= s; b.y *= s; b.z *= s; b.w *= s;
    p[0] = a;
    p[1] = b;
}

extern "C" void kernel_launch(void* const* d_in, const int* in_sizes, int n_in,
                              void* d_out, int out_size, void* d_ws, size_t ws_size,
                              hipStream_t stream) {
    (void)in_sizes; (void)n_in; (void)out_size; (void)ws_size;
    const float* X = (const float*)d_in[0];
    const float* W = (const float*)d_in[1];
    const float* b = (const float*)d_in[2];
    float* out = (float*)d_out;

    // d_ws: two bf16 planes of W, 2 * 5,308,416 * 2 B = 20.25 MB
    unsigned short* Whi = (unsigned short*)d_ws;
    unsigned short* Wlo = Whi + W_ELEMS;

    prep_w<<<dim3(256), dim3(256), 0, stream>>>(W, Whi, Wlo);
    caps_conv_wmma<<<dim3(288), dim3(256), 0, stream>>>(X, Whi, Wlo, b, out);
    caps_squash<<<dim3(2304), dim3(256), 0, stream>>>(out);
}
